// GeneratorNetwork_23510650979053
// MI455X (gfx1250) — compile-verified, matrix operand order checked by reading, not run on hardware
//
#include <hip/hip_runtime.h>

typedef __attribute__((ext_vector_type(16))) __bf16 v16bf;
typedef __attribute__((ext_vector_type(8)))  __bf16 v8bf;
typedef __attribute__((ext_vector_type(8)))  float  v8f;

#define BATCHN  4096
#define RNNW    512
#define GATESN  2048
#define ROWS    16          // batch rows per workgroup (WMMA M)
#define HSTR    520         // padded bf16 LDS row pitch (bank-conflict-free b128 A loads)

__device__ __forceinline__ float sigf(float x) { return 1.f / (1.f + __expf(-x)); }

// A-fragment (16x32 bf16) from LDS h buffer: lane = M (0..15) x half,
// elems j=0..7 -> K = kt*32 + 8*half + j ; j=8..15 -> K = kt*32 + 16 + 8*half + (j-8)
__device__ __forceinline__ v16bf load_afrag(const __bf16* hbuf, int lane, int kt) {
    const int m    = lane & 15;
    const int half = lane >> 4;
    const __bf16* p = hbuf + m * HSTR + kt * 32 + 8 * half;
    v8bf lo = *(const v8bf*)(p);
    v8bf hi = *(const v8bf*)(p + 16);
    return __builtin_shufflevector(lo, hi, 0,1,2,3,4,5,6,7,8,9,10,11,12,13,14,15);
}

// ---------------- one-time kernels ----------------

// idea = tanh(latent @ W_unpack^T + b_unpack), stored bf16 [4096,512]
__global__ void k_idea(const float* __restrict__ latent, const float* __restrict__ Wu,
                       const float* __restrict__ bu, __bf16* __restrict__ idea) {
    int tid = blockIdx.x * blockDim.x + threadIdx.x;
    int b = tid >> 9, c = tid & 511;
    const float* lr = latent + (size_t)b * 512;
    const float* wr = Wu + (size_t)c * 512;
    float acc = bu[c];
#pragma unroll 8
    for (int k = 0; k < 512; ++k) acc += lr[k] * wr[k];
    idea[(size_t)b * 512 + c] = (__bf16)tanhf(acc);
}

// precomp1[b][n] = b_ih1[n]+b_hh1[n] + sum_k bf16(idea[b][k]) * W_ih1[n][k]  (W_ih1 row stride 514)
__global__ void k_precomp(const __bf16* __restrict__ idea, const float* __restrict__ Wih1,
                          const float* __restrict__ bih1, const float* __restrict__ bhh1,
                          float* __restrict__ pre) {
    int tid = blockIdx.x * blockDim.x + threadIdx.x;
    int b = tid >> 11, n = tid & 2047;
    const __bf16* ir = idea + (size_t)b * 512;
    const float*  wr = Wih1 + (size_t)n * 514;
    float acc = bih1[n] + bhh1[n];
#pragma unroll 8
    for (int k = 0; k < 512; ++k) acc += (float)ir[k] * wr[k];
    pre[(size_t)b * GATESN + n] = acc;
}

// pack f32 weight [2048 x 512] (row stride rstride) into bf16 WMMA B-fragment order:
// flat = ((ntile*16 + kt)*32 + lane)*16 + j ; n = ntile*16 + (lane&15), half = lane>>4,
// k = kt*32 + (j<8 ? 8*half + j : 16 + 8*half + (j-8))
__global__ void k_pack(const float* __restrict__ W, int rstride, __bf16* __restrict__ dst) {
    int tid = blockIdx.x * blockDim.x + threadIdx.x;      // 2048*512 threads
    int j = tid & 15, lane = (tid >> 4) & 31, kt = (tid >> 9) & 15, nt = tid >> 13;
    int n = nt * 16 + (lane & 15);
    int half = lane >> 4;
    int k = kt * 32 + ((j < 8) ? (8 * half + j) : (16 + 8 * half + (j - 8)));
    dst[tid] = (__bf16)W[(size_t)n * rstride + k];
}

// wcol[n] = W_ih1[n][512] (the 'angle' input column); bsum2[n] = b_ih2[n]+b_hh2[n]
__global__ void k_misc(const float* __restrict__ Wih1, const float* __restrict__ bih2,
                       const float* __restrict__ bhh2, float* __restrict__ wcol,
                       float* __restrict__ bsum2) {
    int n = blockIdx.x * blockDim.x + threadIdx.x;
    wcol[n]  = Wih1[(size_t)n * 514 + 512];
    bsum2[n] = bih2[n] + bhh2[n];
}

// ---------------- persistent RNN kernel ----------------
__launch_bounds__(256, 1)
__global__ void k_rnn(const float* __restrict__ pre, const __bf16* __restrict__ Whh1,
                      const __bf16* __restrict__ Wih2, const __bf16* __restrict__ Whh2,
                      const float* __restrict__ wcol, const float* __restrict__ bsum2,
                      const float* __restrict__ Wmu, const float* __restrict__ bmu,
                      const float* __restrict__ Wvar, const float* __restrict__ bvar,
                      const float* __restrict__ Wmix, const float* __restrict__ bmix,
                      const float* __restrict__ eps, float* __restrict__ out, int S) {
    __shared__ __bf16 h1s[ROWS * HSTR];
    __shared__ __bf16 h2s[ROWS * HSTR];
    __shared__ float  angles[ROWS];

    const int tid  = threadIdx.x;
    const int lane = tid & 31;
    const int w    = tid >> 5;          // wave 0..7
    const int half = lane >> 4;
    const int nl   = lane & 15;
    const int m0   = blockIdx.x * ROWS; // global batch-row base

    // zero init shared + per-lane state
    for (int i = tid; i < ROWS * HSTR; i += 256) { h1s[i] = (__bf16)0.f; h2s[i] = (__bf16)0.f; }
    if (tid < ROWS) angles[tid] = 0.f;
    float c1[32], c2[32], hn[32];
#pragma unroll
    for (int i = 0; i < 32; ++i) { c1[i] = 0.f; c2[i] = 0.f; }
    __syncthreads();

    for (int s = 0; s < S; ++s) {
        // ---------------- LSTM layer 1 ----------------
#pragma unroll
        for (int p = 0; p < 2; ++p) {
            const int nb = w * 4 + p * 2;
            v8f acc[8];
            // init: precomp1 + angle * wcol  (rank-1 'prev' contribution)
#pragma unroll
            for (int t = 0; t < 8; ++t) {
                const int g = t >> 1, sub = t & 1;
                const int ncol = (g * 32 + nb + sub) * 16 + nl;
                const float wc = wcol[ncol];
#pragma unroll
                for (int v = 0; v < 8; ++v) {
                    const int m = v + 8 * half;
                    acc[t][v] = pre[(size_t)(m0 + m) * GATESN + ncol] + angles[m] * wc;
                }
            }
            // gates1 += h1 @ Whh1^T
#pragma unroll 2
            for (int kt = 0; kt < 16; ++kt) {
                const v16bf a = load_afrag(h1s, lane, kt);
#pragma unroll
                for (int t = 0; t < 8; ++t) {
                    const int g = t >> 1, sub = t & 1;
                    const int nt = g * 32 + nb + sub;
                    const v16bf b = *(const v16bf*)(Whh1 + (((size_t)nt * 16 + kt) * 32 + lane) * 16);
                    acc[t] = __builtin_amdgcn_wmma_f32_16x16x32_bf16(
                        false, a, false, b, (short)0, acc[t], false, false);
                }
            }
            // elementwise cell update (gate order i,f,g,o)
#pragma unroll
            for (int sub = 0; sub < 2; ++sub)
#pragma unroll
                for (int v = 0; v < 8; ++v) {
                    const int ci = p * 16 + sub * 8 + v;
                    const float xi = acc[0 + sub][v], xf = acc[2 + sub][v];
                    const float xg = acc[4 + sub][v], xo = acc[6 + sub][v];
                    float c = sigf(xf) * c1[ci] + sigf(xi) * tanhf(xg);
                    c1[ci] = c;
                    hn[ci] = sigf(xo) * tanhf(c);
                }
        }
        __syncthreads();   // all reads of old h1 done
#pragma unroll
        for (int p = 0; p < 2; ++p)
#pragma unroll
            for (int sub = 0; sub < 2; ++sub)
#pragma unroll
                for (int v = 0; v < 8; ++v)
                    h1s[(v + 8 * half) * HSTR + w * 64 + p * 32 + sub * 16 + nl] =
                        (__bf16)hn[p * 16 + sub * 8 + v];
        __syncthreads();   // new h1 published

        // ---------------- LSTM layer 2 ----------------
#pragma unroll
        for (int p = 0; p < 2; ++p) {
            const int nb = w * 4 + p * 2;
            v8f acc[8];
#pragma unroll
            for (int t = 0; t < 8; ++t) {
                const int g = t >> 1, sub = t & 1;
                const float bs = bsum2[(g * 32 + nb + sub) * 16 + nl];
#pragma unroll
                for (int v = 0; v < 8; ++v) acc[t][v] = bs;
            }
            // gates2 += h1_new @ Wih2^T
#pragma unroll 2
            for (int kt = 0; kt < 16; ++kt) {
                const v16bf a = load_afrag(h1s, lane, kt);
#pragma unroll
                for (int t = 0; t < 8; ++t) {
                    const int g = t >> 1, sub = t & 1;
                    const int nt = g * 32 + nb + sub;
                    const v16bf b = *(const v16bf*)(Wih2 + (((size_t)nt * 16 + kt) * 32 + lane) * 16);
                    acc[t] = __builtin_amdgcn_wmma_f32_16x16x32_bf16(
                        false, a, false, b, (short)0, acc[t], false, false);
                }
            }
            // gates2 += h2_old @ Whh2^T
#pragma unroll 2
            for (int kt = 0; kt < 16; ++kt) {
                const v16bf a = load_afrag(h2s, lane, kt);
#pragma unroll
                for (int t = 0; t < 8; ++t) {
                    const int g = t >> 1, sub = t & 1;
                    const int nt = g * 32 + nb + sub;
                    const v16bf b = *(const v16bf*)(Whh2 + (((size_t)nt * 16 + kt) * 32 + lane) * 16);
                    acc[t] = __builtin_amdgcn_wmma_f32_16x16x32_bf16(
                        false, a, false, b, (short)0, acc[t], false, false);
                }
            }
#pragma unroll
            for (int sub = 0; sub < 2; ++sub)
#pragma unroll
                for (int v = 0; v < 8; ++v) {
                    const int ci = p * 16 + sub * 8 + v;
                    const float xi = acc[0 + sub][v], xf = acc[2 + sub][v];
                    const float xg = acc[4 + sub][v], xo = acc[6 + sub][v];
                    float c = sigf(xf) * c2[ci] + sigf(xi) * tanhf(xg);
                    c2[ci] = c;
                    hn[ci] = sigf(xo) * tanhf(c);
                }
        }
        __syncthreads();   // all reads of old h2 done
#pragma unroll
        for (int p = 0; p < 2; ++p)
#pragma unroll
            for (int sub = 0; sub < 2; ++sub)
#pragma unroll
                for (int v = 0; v < 8; ++v)
                    h2s[(v + 8 * half) * HSTR + w * 64 + p * 32 + sub * 16 + nl] =
                        (__bf16)hn[p * 16 + sub * 8 + v];
        __syncthreads();   // new h2 published

        // ---------------- mixture head: wave w handles rows 2w, 2w+1 ----------------
#pragma unroll 1
        for (int rr = 0; rr < 2; ++rr) {
            const int r = w * 2 + rr;
            float am[5] = {0,0,0,0,0}, av[5] = {0,0,0,0,0}, ax[5] = {0,0,0,0,0};
#pragma unroll 4
            for (int k = lane; k < 512; k += 32) {
                const float hv = (float)h2s[r * HSTR + k];
#pragma unroll
                for (int o = 0; o < 5; ++o) {
                    am[o] += hv * Wmu [o * 512 + k];
                    av[o] += hv * Wvar[o * 512 + k];
                    ax[o] += hv * Wmix[o * 512 + k];
                }
            }
#pragma unroll
            for (int o = 0; o < 5; ++o)
#pragma unroll
                for (int d = 16; d > 0; d >>= 1) {
                    am[o] += __shfl_xor(am[o], d, 32);
                    av[o] += __shfl_xor(av[o], d, 32);
                    ax[o] += __shfl_xor(ax[o], d, 32);
                }
            if (lane == 0) {
                float lg[5], mx = -1e30f;
#pragma unroll
                for (int o = 0; o < 5; ++o) { lg[o] = ax[o] + bmix[o]; mx = fmaxf(mx, lg[o]); }
                float se = 0.f, mu = 0.f, sg = 0.f;
#pragma unroll
                for (int o = 0; o < 5; ++o) {
                    const float rho = __expf(lg[o] - mx);
                    se += rho;
                    mu += rho * (am[o] + bmu[o]);
                    sg += rho * __expf(av[o] + bvar[o]);
                }
                const float inv = 1.f / se;
                const float e = eps[(size_t)s * BATCHN + m0 + r];
                const float ang = tanhf(mu * inv + sg * inv * e);
                angles[r] = ang;
                const size_t oi = ((size_t)(m0 + r) * S + s) * 2;
                out[oi] = ang;
                out[oi + 1] = 0.f;
            }
        }
        __syncthreads();   // angles ready for next step
    }
}

extern "C" void kernel_launch(void* const* d_in, const int* in_sizes, int n_in,
                              void* d_out, int out_size, void* d_ws, size_t ws_size,
                              hipStream_t stream) {
    (void)in_sizes; (void)n_in; (void)ws_size;
    const float* latent = (const float*)d_in[0];
    const float* Wu     = (const float*)d_in[1];
    const float* bu     = (const float*)d_in[2];
    const float* Wih1   = (const float*)d_in[3];   // [2048, 514]
    const float* Whh1f  = (const float*)d_in[4];   // [2048, 512]
    const float* bih1   = (const float*)d_in[5];
    const float* bhh1   = (const float*)d_in[6];
    const float* Wih2f  = (const float*)d_in[7];
    const float* Whh2f  = (const float*)d_in[8];
    const float* bih2   = (const float*)d_in[9];
    const float* bhh2   = (const float*)d_in[10];
    const float* Wmu    = (const float*)d_in[11];
    const float* bmu    = (const float*)d_in[12];
    const float* Wvar   = (const float*)d_in[13];
    const float* bvar   = (const float*)d_in[14];
    const float* Wmix   = (const float*)d_in[15];
    const float* bmix   = (const float*)d_in[16];
    const float* eps    = (const float*)d_in[17];
    float* out = (float*)d_out;
    const int S = out_size / (BATCHN * 2);

    // workspace layout
    char* ws = (char*)d_ws;
    __bf16* idea  = (__bf16*)ws;                                   //  4 MB
    float*  pre   = (float*)(ws + (size_t)4 * 1024 * 1024);        // 32 MB
    __bf16* whh1  = (__bf16*)(ws + (size_t)36 * 1024 * 1024);      //  2 MB each
    __bf16* wih2  = whh1 + (size_t)GATESN * RNNW;
    __bf16* whh2  = wih2 + (size_t)GATESN * RNNW;
    float*  wcol  = (float*)(ws + (size_t)42 * 1024 * 1024);
    float*  bsum2 = wcol + GATESN;

    k_idea   <<<(BATCHN * 512)   / 256, 256, 0, stream>>>(latent, Wu, bu, idea);
    k_pack   <<<(GATESN * RNNW)  / 256, 256, 0, stream>>>(Whh1f, 512, whh1);
    k_pack   <<<(GATESN * RNNW)  / 256, 256, 0, stream>>>(Wih2f, 512, wih2);
    k_pack   <<<(GATESN * RNNW)  / 256, 256, 0, stream>>>(Whh2f, 512, whh2);
    k_misc   <<<GATESN / 256,          256, 0, stream>>>(Wih1, bih2, bhh2, wcol, bsum2);
    k_precomp<<<(BATCHN * GATESN)/ 256, 256, 0, stream>>>(idea, Wih1, bih1, bhh1, pre);
    k_rnn    <<<BATCHN / ROWS,          256, 0, stream>>>(pre, whh1, wih2, whh2, wcol, bsum2,
                                                          Wmu, bmu, Wvar, bvar, Wmix, bmix,
                                                          eps, out, S);
}
